// CausalSelfAttention_80882824118884
// MI455X (gfx1250) — matrix operand order repaired and checked by reading, not compiled, not run on hardware
//
#include <hip/hip_runtime.h>
#include <hip/hip_bf16.h>

#define BB 4
#define TT 2048
#define CC 1024
#define HH 16
#define DD 64
#define MM (BB*TT)   // 8192 rows

typedef __bf16 bf16_t;
typedef __attribute__((ext_vector_type(16))) __bf16 v16bf;
typedef __attribute__((ext_vector_type(8)))  float  v8f;
typedef int v4i_ __attribute__((vector_size(16)));

#if defined(__AMDGCN__) && __has_builtin(__builtin_amdgcn_global_load_async_to_lds_b128)
#define HAVE_ASYNC_LDS 1
#else
#define HAVE_ASYNC_LDS 0
#endif

__device__ __forceinline__ void wait_async_then_barrier() {
#if HAVE_ASYNC_LDS
    asm volatile("s_wait_asynccnt 0x0" ::: "memory");
#endif
    __syncthreads();
}

#if HAVE_ASYNC_LDS
// one 16-byte global -> LDS async copy per lane
__device__ __forceinline__ void async_cp16(const bf16_t* g, bf16_t* l) {
    __builtin_amdgcn_global_load_async_to_lds_b128(
        (__attribute__((address_space(1))) v4i_*)g,
        (__attribute__((address_space(3))) v4i_*)l, 0, 0);
}
#endif

// ---------------------------------------------------------------- converters
__global__ void cvt_f32_bf16(const float* __restrict__ s, bf16_t* __restrict__ d, int n) {
    int i = blockIdx.x * blockDim.x + threadIdx.x;
    int stride = gridDim.x * blockDim.x;
    for (; i < n; i += stride) d[i] = (bf16_t)s[i];
}

// ---------------------------------------------------------------- bf16 GEMM
// C[M,N] = A[M,K](bf16) * B[K,N](bf16) + bias[N](f32)
// block = 256 threads (8 waves); block tile 128(M) x 64(N); K stepped by 32.
// Wave grid 4(M) x 2(N); each wave computes a 32x32 tile = 2x2 WMMA subtiles.
template<int OUT_BF16>
__global__ __launch_bounds__(256) void gemm_bf16_k(
    const bf16_t* __restrict__ A, const bf16_t* __restrict__ Bm,
    const float* __restrict__ bias, void* __restrict__ out,
    int M, int N, int K)
{
    constexpr int AS = 40;                  // LDS row stride: 80B (16B-aligned, bank step 20)
    __shared__ bf16_t Alds[128 * AS];       // [m][k]
    __shared__ bf16_t Btl [64 * AS];        // transposed: [n][k]

    const int tid  = threadIdx.x;
    const int wid  = tid >> 5;
    const int lane = tid & 31;
    const int half = lane >> 4;
    const int l16  = lane & 15;
    const int waveM = wid >> 1;             // 0..3 -> 32-row strip
    const int waveN = wid & 1;              // 0..1 -> 32-col strip

    const int m0 = blockIdx.y * 128;
    const int n0 = blockIdx.x * 64;

    v8f acc[2][2];
    #pragma unroll
    for (int sn = 0; sn < 2; ++sn) {
        float bv = bias[n0 + waveN * 32 + sn * 16 + l16];
        #pragma unroll
        for (int sm = 0; sm < 2; ++sm)
            #pragma unroll
            for (int r = 0; r < 8; ++r) acc[sm][sn][r] = bv;
    }

    for (int k0 = 0; k0 < K; k0 += 32) {
        __syncthreads();                    // WAR: previous tile fully consumed
        // ---- stage A: 128x32 tile (contiguous 16B chunks)
#if HAVE_ASYNC_LDS
        #pragma unroll
        for (int i = 0; i < 2; ++i) {
            int ch = tid + i * 256;         // 512 chunks of 8 elems
            int r = ch >> 2, c = (ch & 3) * 8;
            async_cp16(A + (size_t)(m0 + r) * K + k0 + c, Alds + r * AS + c);
        }
#else
        #pragma unroll
        for (int i = 0; i < 2; ++i) {
            int ch = tid + i * 256;
            int r = ch >> 2, c = (ch & 3) * 8;
            const bf16_t* src = A + (size_t)(m0 + r) * K + k0 + c;
            bf16_t* dst = Alds + r * AS + c;
            #pragma unroll
            for (int j = 0; j < 8; ++j) dst[j] = src[j];
        }
#endif
        // ---- stage B: 32x64 tile transposed into [n][k] (manual: DMA can't transpose)
        {
            int r = tid >> 3;               // k 0..31
            int c = (tid & 7) * 8;          // n
            const bf16_t* src = Bm + (size_t)(k0 + r) * N + n0 + c;
            #pragma unroll
            for (int i = 0; i < 8; ++i) Btl[(c + i) * AS + r] = src[i];
        }
        wait_async_then_barrier();

        // ---- fragments
        const int klo = half * 8;
        v16bf af[2];
        #pragma unroll
        for (int sm = 0; sm < 2; ++sm) {
            const int am = waveM * 32 + sm * 16 + l16;
            #pragma unroll
            for (int j = 0; j < 4; ++j) {
                af[sm][2*j]   = Alds[am * AS + klo + 2*j];
                af[sm][2*j+1] = Alds[am * AS + klo + 2*j + 1];
                af[sm][8+2*j] = Alds[am * AS + 16 + klo + 2*j];
                af[sm][9+2*j] = Alds[am * AS + 16 + klo + 2*j + 1];
            }
        }
        v16bf bfr[2];
        #pragma unroll
        for (int sn = 0; sn < 2; ++sn) {
            const int bn = waveN * 32 + sn * 16 + l16;
            const int kb = half * 8;
            #pragma unroll
            for (int j = 0; j < 4; ++j) {
                bfr[sn][2*j]   = Btl[bn * AS + kb + 2*j];
                bfr[sn][2*j+1] = Btl[bn * AS + kb + 2*j + 1];
                bfr[sn][8+2*j] = Btl[bn * AS + 16 + kb + 2*j];
                bfr[sn][9+2*j] = Btl[bn * AS + 16 + kb + 2*j + 1];
            }
        }
        #pragma unroll
        for (int sm = 0; sm < 2; ++sm)
            #pragma unroll
            for (int sn = 0; sn < 2; ++sn)
                acc[sm][sn] = __builtin_amdgcn_wmma_f32_16x16x32_bf16(
                    false, af[sm], false, bfr[sn], (short)0, acc[sm][sn], false, false);
    }

    #pragma unroll
    for (int sm = 0; sm < 2; ++sm)
        #pragma unroll
        for (int sn = 0; sn < 2; ++sn) {
            int gn = n0 + waveN * 32 + sn * 16 + l16;
            #pragma unroll
            for (int r = 0; r < 8; ++r) {
                int gm = m0 + waveM * 32 + sm * 16 + r + 8 * half;
                if (OUT_BF16) ((bf16_t*)out)[(size_t)gm * N + gn] = (bf16_t)acc[sm][sn][r];
                else          ((float*) out)[(size_t)gm * N + gn] = acc[sm][sn][r];
            }
        }
}

// ---------------------------------------------------------------- flash attention
// qkv: [B*T, 3C] bf16 ; y: [B*T, C] bf16
// block = 128 threads (4 waves) handles 64 queries of one (b,h);
// each wave owns 16 query rows, keys processed in causal 64-wide blocks.
__global__ __launch_bounds__(128) void attn_k(const bf16_t* __restrict__ qkv,
                                              bf16_t* __restrict__ y)
{
    constexpr int KS = 72;                       // 144B stride (16B-aligned, bank step 36)
    __shared__ bf16_t Klds[64 * KS];             // [key][d]
    __shared__ bf16_t Vt  [64 * KS];             // transposed: [d][key]
    __shared__ bf16_t Plds[4 * 16 * KS];         // per-wave 16x64 P tile

    const int tid  = threadIdx.x;
    const int wid  = tid >> 5;
    const int lane = tid & 31;
    const int half = lane >> 4;
    const int l16  = lane & 15;

    const int qt = blockIdx.x;                   // query tile (0..T/64-1)
    const int bh = blockIdx.y;
    const int b = bh / HH, h = bh % HH;

    const size_t rs = 3 * CC;                    // qkv row stride
    const bf16_t* qb  = qkv + (size_t)b * TT * rs + h * DD;
    const bf16_t* kbp = qb + CC;
    const bf16_t* vbp = qb + 2 * CC;

    // load Q fragments: two K=32 chunks covering D=64 (A-matrix layout)
    v16bf qf[2];
    {
        const int qrow = qt * 64 + wid * 16 + l16;
        const bf16_t* qp = qb + (size_t)qrow * rs;
        const int klo = half * 8;
        #pragma unroll
        for (int c = 0; c < 2; ++c)
            #pragma unroll
            for (int j = 0; j < 4; ++j) {
                qf[c][2*j]   = qp[c*32 + klo + 2*j];
                qf[c][2*j+1] = qp[c*32 + klo + 2*j + 1];
                qf[c][8+2*j] = qp[c*32 + 16 + klo + 2*j];
                qf[c][9+2*j] = qp[c*32 + 16 + klo + 2*j + 1];
            }
    }

    float m_i[8], l_i[8];
    v8f o[4];
    #pragma unroll
    for (int r = 0; r < 8; ++r) { m_i[r] = -3.0e38f; l_i[r] = 0.0f; }
    #pragma unroll
    for (int dt = 0; dt < 4; ++dt)
        #pragma unroll
        for (int r = 0; r < 8; ++r) o[dt][r] = 0.0f;

    const float scale = 0.125f;                  // 1/sqrt(64)

    for (int kb = 0; kb <= qt; ++kb) {
        __syncthreads();
        // ---- stage K via async DMA (natural [key][d] layout, 16B chunks)
#if HAVE_ASYNC_LDS
        #pragma unroll
        for (int i = 0; i < 4; ++i) {
            int ch = tid + i * 128;              // 512 chunks of 8 elems
            int r = ch >> 3, c = (ch & 7) * 8;
            async_cp16(kbp + (size_t)(kb * 64 + r) * rs + c, Klds + r * KS + c);
        }
#else
        for (int i = tid; i < 64 * 64; i += 128) {
            int r = i >> 6, c = i & 63;
            Klds[r * KS + c] = kbp[(size_t)(kb * 64 + r) * rs + c];
        }
#endif
        // ---- stage V transposed (manual)
        for (int i = tid; i < 64 * 64; i += 128) {
            int r = i >> 6, c = i & 63;
            Vt[c * KS + r] = vbp[(size_t)(kb * 64 + r) * rs + c];
        }
        wait_async_then_barrier();

        // S = Q K^T  (4 key sub-tiles of 16, K-dim = 64 in two WMMAs)
        v8f s[4];
        #pragma unroll
        for (int nt = 0; nt < 4; ++nt) {
            v8f a;
            #pragma unroll
            for (int r = 0; r < 8; ++r) a[r] = 0.0f;
            #pragma unroll
            for (int c = 0; c < 2; ++c) {
                const int kd = c * 32 + half * 8;
                const int key = nt * 16 + l16;
                v16bf bfv;
                #pragma unroll
                for (int j = 0; j < 4; ++j) {
                    bfv[2*j]   = Klds[key * KS + kd + 2*j];
                    bfv[2*j+1] = Klds[key * KS + kd + 2*j + 1];
                    bfv[8+2*j] = Klds[key * KS + kd + 16 + 2*j];
                    bfv[9+2*j] = Klds[key * KS + kd + 16 + 2*j + 1];
                }
                a = __builtin_amdgcn_wmma_f32_16x16x32_bf16(
                    false, qf[c], false, bfv, (short)0, a, false, false);
            }
            s[nt] = a;
        }

        // scale + causal mask (diagonal block only) + online softmax
        const bool diag = (kb == qt);
        #pragma unroll
        for (int r = 0; r < 8; ++r) {
            const int q_g = qt * 64 + wid * 16 + r + 8 * half;
            float tmax = -3.0e38f;
            #pragma unroll
            for (int nt = 0; nt < 4; ++nt) {
                float v = s[nt][r] * scale;
                if (diag) {
                    int k_g = kb * 64 + nt * 16 + l16;
                    if (k_g > q_g) v = -3.0e38f;
                }
                s[nt][r] = v;
                tmax = fmaxf(tmax, v);
            }
            #pragma unroll
            for (int off = 1; off < 16; off <<= 1)
                tmax = fmaxf(tmax, __shfl_xor(tmax, off, 32));
            const float nm = fmaxf(m_i[r], tmax);
            const float corr = __expf(m_i[r] - nm);
            m_i[r] = nm;
            float psum = 0.0f;
            #pragma unroll
            for (int nt = 0; nt < 4; ++nt) {
                float p = __expf(s[nt][r] - nm);
                s[nt][r] = p;
                psum += p;
            }
            #pragma unroll
            for (int off = 1; off < 16; off <<= 1)
                psum += __shfl_xor(psum, off, 32);
            l_i[r] = l_i[r] * corr + psum;
            #pragma unroll
            for (int dt = 0; dt < 4; ++dt) o[dt][r] *= corr;
            // P (C-layout) -> per-wave LDS for A-layout reload
            #pragma unroll
            for (int nt = 0; nt < 4; ++nt)
                Plds[(wid * 16 + r + 8 * half) * KS + nt * 16 + l16] = (bf16_t)s[nt][r];
        }

        // reload P as A-fragments (same wave: DS ops in-order, no barrier needed)
        v16bf pf[2];
        {
            const int klo = half * 8;
            const bf16_t* pr = Plds + (size_t)(wid * 16 + l16) * KS;
            #pragma unroll
            for (int c = 0; c < 2; ++c)
                #pragma unroll
                for (int j = 0; j < 4; ++j) {
                    pf[c][2*j]   = pr[c*32 + klo + 2*j];
                    pf[c][2*j+1] = pr[c*32 + klo + 2*j + 1];
                    pf[c][8+2*j] = pr[c*32 + 16 + klo + 2*j];
                    pf[c][9+2*j] = pr[c*32 + 16 + klo + 2*j + 1];
                }
        }
        // O += P V
        #pragma unroll
        for (int dt = 0; dt < 4; ++dt) {
            #pragma unroll
            for (int c = 0; c < 2; ++c) {
                const int kd = c * 32 + half * 8;
                const int d = dt * 16 + l16;
                v16bf bfv;
                #pragma unroll
                for (int j = 0; j < 4; ++j) {
                    bfv[2*j]   = Vt[d * KS + kd + 2*j];
                    bfv[2*j+1] = Vt[d * KS + kd + 2*j + 1];
                    bfv[8+2*j] = Vt[d * KS + kd + 16 + 2*j];
                    bfv[9+2*j] = Vt[d * KS + kd + 16 + 2*j + 1];
                }
                o[dt] = __builtin_amdgcn_wmma_f32_16x16x32_bf16(
                    false, pf[c], false, bfv, (short)0, o[dt], false, false);
            }
        }
    }

    // normalize and store y (bf16) into [B*T, C] at head offset
    #pragma unroll
    for (int r = 0; r < 8; ++r) {
        const float inv = 1.0f / l_i[r];
        const int t = qt * 64 + wid * 16 + r + 8 * half;
        const size_t row = ((size_t)b * TT + t) * CC + h * DD;
        #pragma unroll
        for (int dt = 0; dt < 4; ++dt)
            y[row + dt * 16 + l16] = (bf16_t)(o[dt][r] * inv);
    }
}

// ---------------------------------------------------------------- launcher
extern "C" void kernel_launch(void* const* d_in, const int* in_sizes, int n_in,
                              void* d_out, int out_size, void* d_ws, size_t ws_size,
                              hipStream_t stream) {
    const float* x      = (const float*)d_in[0];
    const float* W_attn = (const float*)d_in[1];
    const float* b_attn = (const float*)d_in[2];
    const float* W_proj = (const float*)d_in[3];
    const float* b_proj = (const float*)d_in[4];

    char* p = (char*)d_ws;
    bf16_t* xb   = (bf16_t*)p; p += (size_t)MM * CC * 2;
    bf16_t* Wab  = (bf16_t*)p; p += (size_t)CC * 3 * CC * 2;
    bf16_t* Wpb  = (bf16_t*)p; p += (size_t)CC * CC * 2;
    bf16_t* qkvb = (bf16_t*)p; p += (size_t)MM * 3 * CC * 2;
    bf16_t* yb   = (bf16_t*)p; p += (size_t)MM * CC * 2;

    cvt_f32_bf16<<<2048, 256, 0, stream>>>(x,      xb,  MM * CC);
    cvt_f32_bf16<<<2048, 256, 0, stream>>>(W_attn, Wab, CC * 3 * CC);
    cvt_f32_bf16<<<1024, 256, 0, stream>>>(W_proj, Wpb, CC * CC);

    // QKV: [8192,1024] x [1024,3072] -> bf16 qkv
    gemm_bf16_k<1><<<dim3(3 * CC / 64, MM / 128), 256, 0, stream>>>(
        xb, Wab, b_attn, qkvb, MM, 3 * CC, CC);

    // fused causal attention -> bf16 y
    attn_k<<<dim3(TT / 64, BB * HH), 128, 0, stream>>>(qkvb, yb);

    // projection: [8192,1024] x [1024,1024] -> fp32 out
    gemm_bf16_k<0><<<dim3(CC / 64, MM / 128), 256, 0, stream>>>(
        yb, Wpb, b_proj, (float*)d_out, MM, CC, CC);
}
